// ImprovedTransformerAutoencoder_53944789237842
// MI455X (gfx1250) — matrix operand order repaired and checked by reading, not compiled, not run on hardware
//
#include <hip/hip_runtime.h>
#include <hip/hip_bf16.h>

// ---------------------------------------------------------------------------
// DeepSeek-sparse-attention transformer layer for gfx1250 (MI455X, wave32).
// GEMMs: v_wmma_f32_16x16x32_f16 with double-buffered LDS B tiles filled by
// GLOBAL_LOAD_ASYNC_TO_LDS_B128 (ASYNCcnt) so copies overlap WMMA compute.
// Attention exploits the top-k mask (softmax over 64 gathered keys only).
// ---------------------------------------------------------------------------

typedef __attribute__((ext_vector_type(16))) _Float16 v16h;
typedef __attribute__((ext_vector_type(8)))  float    v8f;

union frag16 { v16h v; unsigned int u[8]; };

// ---- 16-bit A-matrix fragment (16x32, MxK), contiguous k per lane half (this code's order) ----
// lane L(0..15): M=L, K in {0..15}; lane L+16: M=L, K in {16..31}
__device__ inline v16h load_a_frag(const _Float16* __restrict__ A, int ldk,
                                   int row, int k0, int lane) {
  frag16 f;
  int g = (lane >> 4) & 1;
  const unsigned int* base =
      reinterpret_cast<const unsigned int*>(A + (size_t)row * ldk + k0);
#pragma unroll
  for (int v = 0; v < 8; ++v) {
    int kk = v * 2 + g * 16;
    f.u[v] = base[kk >> 1];
  }
  return f.v;
}

// ---- 16-bit B-matrix fragment (32x16, KxN) from LDS tile Bs[n][kk] ----
// lane: N = lane%16; element j: K = j + 16*(lane/16)
__device__ inline v16h load_b_frag_lds(const _Float16* __restrict__ Bs,
                                       int ntile, int lane) {
  frag16 f;
  int g = (lane >> 4) & 1;
  int n = ntile * 16 + (lane & 15);
  const unsigned int* base =
      reinterpret_cast<const unsigned int*>(Bs + n * 32 + g * 16);
#pragma unroll
  for (int v = 0; v < 8; ++v) f.u[v] = base[v];
  return f.v;
}

// ---- async DMA of one wave's 8-row slice of a 64x32 f16 B tile into LDS ----
// lane -> row = wave*8 + lane/4, seg = lane%4 (16 bytes each); one
// global_load_async_to_lds_b128 per wave covers 512B; tracked by ASYNCcnt.
__device__ inline void async_copy_b_tile(const _Float16* __restrict__ W,
                                         _Float16* bufp, int n0, int N, int K,
                                         int k0, int wave, int lane) {
  int row = wave * 8 + (lane >> 2);
  int seg = lane & 3;
  int n = n0 + row;
  int nn = (n < N) ? n : 0;  // clamp: garbage feeds only discarded columns
  unsigned lds_off = (unsigned)(uintptr_t)(bufp + row * 32 + seg * 8);
  const _Float16* g = W + (size_t)nn * K + k0 + seg * 8;
  asm volatile("global_load_async_to_lds_b128 %0, %1, off"
               :: "v"(lds_off), "v"(g) : "memory");
}

// ---------------------------------------------------------------------------
// f32 -> f16 conversion (weights, activation packing)
// ---------------------------------------------------------------------------
__global__ __launch_bounds__(256) void k_f2h(const float* __restrict__ in,
                                             _Float16* __restrict__ out, int n) {
  int i = blockIdx.x * 256 + threadIdx.x;
  if (i < n) out[i] = (_Float16)in[i];
}

// ---------------------------------------------------------------------------
// LayerNorm over D=512, one block per row, writes f16 for WMMA consumption
// ---------------------------------------------------------------------------
__global__ __launch_bounds__(256) void k_layernorm(const float* __restrict__ x,
                                                   const float* __restrict__ gamma,
                                                   const float* __restrict__ beta,
                                                   _Float16* __restrict__ outh) {
  const int D = 512;
  int row = blockIdx.x;
  int tid = threadIdx.x;
  const float* xr = x + (size_t)row * D;
  float v0 = xr[tid], v1 = xr[tid + 256];
  float s = v0 + v1, s2 = v0 * v0 + v1 * v1;
#pragma unroll
  for (int o = 16; o > 0; o >>= 1) {
    s  += __shfl_xor(s,  o, 32);
    s2 += __shfl_xor(s2, o, 32);
  }
  __shared__ float ssum[8], ssq[8], mb[2];
  if ((tid & 31) == 0) { ssum[tid >> 5] = s; ssq[tid >> 5] = s2; }
  __syncthreads();
  if (tid == 0) {
    float ts = 0.f, tq = 0.f;
#pragma unroll
    for (int i = 0; i < 8; ++i) { ts += ssum[i]; tq += ssq[i]; }
    float mu = ts / D;
    float var = tq / D - mu * mu;
    mb[0] = mu; mb[1] = rsqrtf(var + 1e-5f);
  }
  __syncthreads();
  float mu = mb[0], inv = mb[1];
  outh[(size_t)row * D + tid]       = (_Float16)((v0 - mu) * inv * gamma[tid] + beta[tid]);
  outh[(size_t)row * D + tid + 256] = (_Float16)((v1 - mu) * inv * gamma[tid + 256] + beta[tid + 256]);
}

// ---------------------------------------------------------------------------
// GEMM: C[M,N] = A[M,K](f16) @ W[N,K](f16)^T + bias (+resid) [gelu]
// block = 256 thr = 8 waves; block tile 128(M) x 64(N).
// B tiles double-buffered in LDS, filled by async-to-LDS DMA; A fragments
// software-pipelined one step ahead in registers.
// ---------------------------------------------------------------------------
__global__ __launch_bounds__(256) void k_gemm(const _Float16* __restrict__ A,
                                              const _Float16* __restrict__ W,
                                              const float* __restrict__ bias,
                                              const float* __restrict__ resid,
                                              float* __restrict__ Cf,
                                              _Float16* __restrict__ Ch,
                                              int M, int N, int K, int act_gelu) {
  __shared__ __align__(16) _Float16 Bs[2][64 * 32];
  int tid  = threadIdx.x;
  int wave = tid >> 5, lane = tid & 31;
  int n0 = blockIdx.x * 64;
  int m0 = blockIdx.y * 128 + wave * 16;

  v8f acc[4];
#pragma unroll
  for (int i = 0; i < 4; ++i)
#pragma unroll
    for (int j = 0; j < 8; ++j) acc[i][j] = 0.f;

  int nsteps = K >> 5;
  int arow = m0 + (lane & 15);

  // prologue: DMA tile 0, prefetch A fragment 0
  async_copy_b_tile(W, &Bs[0][0], n0, N, K, 0, wave, lane);
  v16h a_cur = load_a_frag(A, K, arow, 0, lane);

  for (int ks = 0; ks < nsteps; ++ks) {
    v16h a_next = a_cur;
    if (ks + 1 < nsteps) {
      // DMA next tile into the other buffer; it flies under this iteration's
      // WMMAs. Then wait only for the *current* tile (asynccnt <= 1).
      async_copy_b_tile(W, &Bs[(ks + 1) & 1][0], n0, N, K, (ks + 1) << 5,
                        wave, lane);
      a_next = load_a_frag(A, K, arow, (ks + 1) << 5, lane);
      asm volatile("s_wait_asynccnt 0x1" ::: "memory");
    } else {
      asm volatile("s_wait_asynccnt 0x0" ::: "memory");
    }
    __syncthreads();  // all waves' DMA slices for tile ks have landed

    const _Float16* bufp = &Bs[ks & 1][0];
#pragma unroll
    for (int nt = 0; nt < 4; ++nt) {
      v16h bf = load_b_frag_lds(bufp, nt, lane);
      acc[nt] = __builtin_amdgcn_wmma_f32_16x16x32_f16(
          false, a_cur, false, bf, (short)0, acc[nt], false, false);
    }
    __syncthreads();  // reads done before this buffer is DMA'd again (ks+2)
    a_cur = a_next;
  }

  // epilogue: C layout -> lane: N=lane%16, VGPR v: M = v + 8*(lane/16)
  int g = lane >> 4;
#pragma unroll
  for (int nt = 0; nt < 4; ++nt) {
    int n = n0 + nt * 16 + (lane & 15);
    if (n >= N) continue;
    float bv = bias ? bias[n] : 0.f;
#pragma unroll
    for (int v = 0; v < 8; ++v) {
      int m = m0 + v + g * 8;
      float val = acc[nt][v] + bv;
      if (resid) val += resid[(size_t)m * N + n];
      if (act_gelu) val = 0.5f * val * (1.0f + erff(val * 0.70710678118654752f));
      if (Cf) Cf[(size_t)m * N + n] = val;
      if (Ch) Ch[(size_t)m * N + n] = (_Float16)val;
    }
  }
}

// ---------------------------------------------------------------------------
// w_idx projection: N=4, done with VALU dot products (tiny)
// ---------------------------------------------------------------------------
__global__ __launch_bounds__(128) void k_wproj(const _Float16* __restrict__ xnh,
                                               const float* __restrict__ iw_w,
                                               const float* __restrict__ iw_b,
                                               float* __restrict__ widx) {
  int m = blockIdx.x;
  int h = threadIdx.x >> 5, lane = threadIdx.x & 31;
  const _Float16* xr = xnh + (size_t)m * 512;
  const float* wr = iw_w + h * 512;
  float s = 0.f;
  for (int i = lane; i < 512; i += 32) s += (float)xr[i] * wr[i];
#pragma unroll
  for (int o = 16; o > 0; o >>= 1) s += __shfl_xor(s, o, 32);
  if (lane == 0) widx[(size_t)m * 4 + h] = s + iw_b[h];
}

// ---------------------------------------------------------------------------
// Indexer scores: fused per-head WMMA (K=32) + relu + head-weight sum.
// grid: (S/128, S/16, B); wave -> one 16x16 (t,s) tile, 4 WMMAs (heads).
// ---------------------------------------------------------------------------
__global__ __launch_bounds__(256) void k_score(const _Float16* __restrict__ qidx,
                                               const _Float16* __restrict__ kidx,
                                               const float* __restrict__ widx,
                                               float* __restrict__ scores,
                                               int S, int Bsz) {
  int lane = threadIdx.x & 31, wave = threadIdx.x >> 5;
  int b  = blockIdx.z;
  int t0 = blockIdx.y * 16;
  int s0 = blockIdx.x * 128 + wave * 16;
  int g  = lane >> 4;

  v8f acc[4];
#pragma unroll
  for (int i = 0; i < 4; ++i)
#pragma unroll
    for (int j = 0; j < 8; ++j) acc[i][j] = 0.f;

  int scol = s0 + (lane & 15);
#pragma unroll
  for (int h = 0; h < 4; ++h) {
    // A: q_idx[b, t, h, :]  flat [(t*B+b)*128 + h*32 + k]
    v16h a = load_a_frag(qidx + (size_t)b * 128 + h * 32, Bsz * 128,
                         t0 + (lane & 15), 0, lane);
    // B: k_idx[b, s, :]  flat [(s*B+b)*32 + k]
    frag16 fb;
    const unsigned int* bb = reinterpret_cast<const unsigned int*>(
        kidx + ((size_t)scol * Bsz + b) * 32 + g * 16);
#pragma unroll
    for (int v = 0; v < 8; ++v) fb.u[v] = bb[v];
    acc[h] = __builtin_amdgcn_wmma_f32_16x16x32_f16(
        false, a, false, fb.v, (short)0, acc[h], false, false);
  }

#pragma unroll
  for (int v = 0; v < 8; ++v) {
    int t = t0 + v + g * 8;
    const float4 w4 =
        *reinterpret_cast<const float4*>(widx + ((size_t)t * Bsz + b) * 4);
    float r = w4.x * fmaxf(acc[0][v], 0.f) + w4.y * fmaxf(acc[1][v], 0.f) +
              w4.z * fmaxf(acc[2][v], 0.f) + w4.w * fmaxf(acc[3][v], 0.f);
    scores[((size_t)b * S + t) * S + scol] = r;
  }
}

// ---------------------------------------------------------------------------
// top-64 per (b,t) row of S=2048; iterative argmax over an LDS-resident row.
// Tie-break to smaller index (matches stable descending sort of top_k).
// ---------------------------------------------------------------------------
__global__ __launch_bounds__(256) void k_topk(const float* __restrict__ scores,
                                              int* __restrict__ topk,
                                              int S, int Kt) {
  __shared__ float sc[2048];
  __shared__ float rv[8];
  __shared__ int   ri[8];
  int row = blockIdx.x;  // b*S + t
  int tid = threadIdx.x;
  const float* srow = scores + (size_t)row * S;
  for (int i = tid; i < S; i += 256) sc[i] = srow[i];
  __syncthreads();
  for (int it = 0; it < Kt; ++it) {
    float bv = -3.4e38f; int bi = S;
    for (int i = tid; i < S; i += 256) {
      float v = sc[i];
      if (v > bv || (v == bv && i < bi)) { bv = v; bi = i; }
    }
#pragma unroll
    for (int o = 16; o > 0; o >>= 1) {
      float ov = __shfl_xor(bv, o, 32);
      int   oi = __shfl_xor(bi, o, 32);
      if (ov > bv || (ov == bv && oi < bi)) { bv = ov; bi = oi; }
    }
    if ((tid & 31) == 0) { rv[tid >> 5] = bv; ri[tid >> 5] = bi; }
    __syncthreads();
    if (tid == 0) {
      float fv = rv[0]; int fi = ri[0];
#pragma unroll
      for (int w = 1; w < 8; ++w)
        if (rv[w] > fv || (rv[w] == fv && ri[w] < fi)) { fv = rv[w]; fi = ri[w]; }
      topk[(size_t)row * Kt + it] = fi;
      sc[fi] = -3.4e38f;
    }
    __syncthreads();
  }
}

// ---------------------------------------------------------------------------
// Sparse attention: softmax over the 64 gathered keys only (exact under the
// additive -inf mask). One block per (b,t); all 8 heads share the key set.
// Writes the (S,B,D)-layout context in f16 for the Wo GEMM.
// ---------------------------------------------------------------------------
__global__ __launch_bounds__(256) void k_attn(const float* __restrict__ Q,
                                              const float* __restrict__ Kf,
                                              const float* __restrict__ Vf,
                                              const int* __restrict__ topk,
                                              _Float16* __restrict__ saH,
                                              int S, int Bsz) {
  __shared__ int   idx[64];
  __shared__ float qs[512];
  __shared__ float lg[512];
  __shared__ float pr[512];
  int blk = blockIdx.x;
  int b = blk / S, t = blk % S;
  int tid = threadIdx.x;
  if (tid < 64) idx[tid] = topk[(size_t)blk * 64 + tid];
  size_t qoff = ((size_t)t * Bsz + b) * 512;
  qs[tid] = Q[qoff + tid];
  qs[tid + 256] = Q[qoff + tid + 256];
  __syncthreads();

  // logits (scaled by 1/sqrt(64))
  for (int p = tid; p < 512; p += 256) {
    int h = p >> 6, j = p & 63;
    int s = idx[j];
    const float* kr = Kf + ((size_t)s * Bsz + b) * 512 + h * 64;
    const float* qr = qs + h * 64;
    float d = 0.f;
#pragma unroll
    for (int e = 0; e < 64; ++e) d += qr[e] * kr[e];
    lg[p] = d * 0.125f;
  }
  __syncthreads();

  // softmax per head: one wave per head, 2 elems per lane
  {
    int wave = tid >> 5, lane = tid & 31;
    float v0 = lg[wave * 64 + lane], v1 = lg[wave * 64 + 32 + lane];
    float mx = fmaxf(v0, v1);
#pragma unroll
    for (int o = 16; o > 0; o >>= 1) mx = fmaxf(mx, __shfl_xor(mx, o, 32));
    float e0 = expf(v0 - mx), e1 = expf(v1 - mx);
    float sm = e0 + e1;
#pragma unroll
    for (int o = 16; o > 0; o >>= 1) sm += __shfl_xor(sm, o, 32);
    float inv = 1.0f / sm;
    pr[wave * 64 + lane] = e0 * inv;
    pr[wave * 64 + 32 + lane] = e1 * inv;
  }
  __syncthreads();

  // context = P @ gathered V
  for (int p = tid; p < 512; p += 256) {
    int h = p >> 6, d = p & 63;
    float o = 0.f;
#pragma unroll
    for (int j = 0; j < 64; ++j) {
      int s = idx[j];
      o += pr[h * 64 + j] * Vf[((size_t)s * Bsz + b) * 512 + h * 64 + d];
    }
    saH[qoff + p] = (_Float16)o;
  }
}

// ---------------------------------------------------------------------------
// Host-side orchestration
// ---------------------------------------------------------------------------
extern "C" void kernel_launch(void* const* d_in, const int* in_sizes, int n_in,
                              void* d_out, int out_size, void* d_ws, size_t ws_size,
                              hipStream_t stream) {
  const int S = 2048, B = 4, D = 512, FF = 2048, Kt = 64;
  const int M = S * B;  // 8192 token rows, m = s*B + b

  const float* src  = (const float*)d_in[0];
  const float* Wq   = (const float*)d_in[1];  const float* bq = (const float*)d_in[2];
  const float* Wk   = (const float*)d_in[3];  const float* bk = (const float*)d_in[4];
  const float* Wv   = (const float*)d_in[5];  const float* bv = (const float*)d_in[6];
  const float* Wo   = (const float*)d_in[7];  const float* bo = (const float*)d_in[8];
  const float* iq_w = (const float*)d_in[9];  const float* iq_b = (const float*)d_in[10];
  const float* ik_w = (const float*)d_in[11]; const float* ik_b = (const float*)d_in[12];
  const float* iw_w = (const float*)d_in[13]; const float* iw_b = (const float*)d_in[14];
  const float* ln1g = (const float*)d_in[15]; const float* ln1b = (const float*)d_in[16];
  const float* ln2g = (const float*)d_in[17]; const float* ln2b = (const float*)d_in[18];
  const float* W1   = (const float*)d_in[19]; const float* b1 = (const float*)d_in[20];
  const float* W2   = (const float*)d_in[21]; const float* b2 = (const float*)d_in[22];
  float* out = (float*)d_out;

  char* w = (char*)d_ws;
  auto alloc = [&](size_t bytes) -> char* {
    char* p = w;
    w += (bytes + 255) & ~(size_t)255;
    return p;
  };

  _Float16* WqH   = (_Float16*)alloc((size_t)D * D * 2);
  _Float16* WkH   = (_Float16*)alloc((size_t)D * D * 2);
  _Float16* WvH   = (_Float16*)alloc((size_t)D * D * 2);
  _Float16* WoH   = (_Float16*)alloc((size_t)D * D * 2);
  _Float16* W1H   = (_Float16*)alloc((size_t)FF * D * 2);
  _Float16* W2H   = (_Float16*)alloc((size_t)D * FF * 2);
  _Float16* iqH   = (_Float16*)alloc((size_t)128 * D * 2);
  _Float16* ikH   = (_Float16*)alloc((size_t)32 * D * 2);
  _Float16* xnH   = (_Float16*)alloc((size_t)M * D * 2);
  _Float16* qidxH = (_Float16*)alloc((size_t)M * 128 * 2);
  _Float16* kidxH = (_Float16*)alloc((size_t)M * 32 * 2);
  float*    widx  = (float*)   alloc((size_t)M * 4 * 4);
  float*    Qf    = (float*)   alloc((size_t)M * D * 4);
  float*    Kf    = (float*)   alloc((size_t)M * D * 4);
  float*    Vf    = (float*)   alloc((size_t)M * D * 4);
  float*    scores= (float*)   alloc((size_t)B * S * S * 4);   // 64 MB
  int*      topk  = (int*)     alloc((size_t)M * Kt * 4);
  _Float16* saH   = (_Float16*)alloc((size_t)M * D * 2);
  float*    xres  = (float*)   alloc((size_t)M * D * 4);
  _Float16* xn2H  = (_Float16*)alloc((size_t)M * D * 2);
  // h1H (M x FF f16 = 32 MB) overlaps the dead 64 MB score buffer
  _Float16* h1H   = (_Float16*)scores;

  auto conv = [&](const float* in, _Float16* outp, int n) {
    k_f2h<<<(n + 255) / 256, 256, 0, stream>>>(in, outp, n);
  };
  conv(Wq, WqH, D * D);   conv(Wk, WkH, D * D);
  conv(Wv, WvH, D * D);   conv(Wo, WoH, D * D);
  conv(W1, W1H, FF * D);  conv(W2, W2H, D * FF);
  conv(iq_w, iqH, 128 * D); conv(ik_w, ikH, 32 * D);

  // LN1 -> xn (f16)
  k_layernorm<<<M, 256, 0, stream>>>(src, ln1g, ln1b, xnH);

  // indexer projections (WMMA GEMMs) + w projection (VALU)
  k_gemm<<<dim3(2, M / 128), 256, 0, stream>>>(xnH, iqH, iq_b, nullptr,
                                               nullptr, qidxH, M, 128, D, 0);
  k_gemm<<<dim3(1, M / 128), 256, 0, stream>>>(xnH, ikH, ik_b, nullptr,
                                               nullptr, kidxH, M, 32, D, 0);
  k_wproj<<<M, 128, 0, stream>>>(xnH, iw_w, iw_b, widx);

  // fused indexer scores + top-64
  k_score<<<dim3(S / 128, S / 16, B), 256, 0, stream>>>(qidxH, kidxH, widx,
                                                        scores, S, B);
  k_topk<<<B * S, 256, 0, stream>>>(scores, topk, S, Kt);

  // QKV projections
  k_gemm<<<dim3(D / 64, M / 128), 256, 0, stream>>>(xnH, WqH, bq, nullptr,
                                                    Qf, nullptr, M, D, D, 0);
  k_gemm<<<dim3(D / 64, M / 128), 256, 0, stream>>>(xnH, WkH, bk, nullptr,
                                                    Kf, nullptr, M, D, D, 0);
  k_gemm<<<dim3(D / 64, M / 128), 256, 0, stream>>>(xnH, WvH, bv, nullptr,
                                                    Vf, nullptr, M, D, D, 0);

  // sparse attention over the gathered top-64 keys
  k_attn<<<B * S, 256, 0, stream>>>(Qf, Kf, Vf, topk, saH, S, B);

  // output projection + residual: x = src + sa @ Wo^T + bo
  k_gemm<<<dim3(D / 64, M / 128), 256, 0, stream>>>(saH, WoH, bo, src,
                                                    xres, nullptr, M, D, D, 0);

  // LN2 -> xn2 (f16)
  k_layernorm<<<M, 256, 0, stream>>>(xres, ln2g, ln2b, xn2H);

  // FFN: h1 = gelu(xn2 @ W1^T + b1) (f16), out = x + h1 @ W2^T + b2
  k_gemm<<<dim3(FF / 64, M / 128), 256, 0, stream>>>(xn2H, W1H, b1, nullptr,
                                                     nullptr, h1H, M, FF, D, 1);
  k_gemm<<<dim3(D / 64, M / 128), 256, 0, stream>>>(h1H, W2H, b2, xres,
                                                    out, nullptr, M, D, FF, 0);

  (void)in_sizes; (void)n_in; (void)out_size; (void)ws_size;
}
